// AxialAttentionBlock_63402307224230
// MI455X (gfx1250) — compile-verified, matrix operand order checked by reading, not run on hardware
//
#include <hip/hip_runtime.h>
#include <hip/hip_bf16.h>

typedef __attribute__((ext_vector_type(16))) _Float16 v16h;
typedef __attribute__((ext_vector_type(8)))  float    v8f;
typedef __attribute__((ext_vector_type(4)))  unsigned int u32x4;
typedef __attribute__((ext_vector_type(8)))  int      i32x8;
typedef __attribute__((ext_vector_type(4)))  int      i32x4;

namespace {
constexpr int kB  = 2;
constexpr int kC  = 768;
constexpr int kH  = 128;
constexpr int kW  = 128;
constexpr int kHW = kH * kW;      // 16384
constexpr int kHE = 12;
constexpr int kHD = 64;
constexpr int kC3 = 3 * kC;       // 2304
constexpr int kC4 = 4 * kC;       // 3072
constexpr int kM  = kB * kHW;     // 32768
constexpr float kEpsIn = 1e-8f;
constexpr float kEpsLn = 1e-5f;

// ---- workspace arena (bytes) ----
constexpr size_t SZ_XN   = (size_t)kM * kC * 2;    // f16 [M,C]
constexpr size_t SZ_QKV  = (size_t)kM * kC3 * 4;   // f32 [M,3C]
constexpr size_t SZ_Q16  = (size_t)kM * kC * 2;    // f16 [B,he,H,W,hd]
constexpr size_t OFF_XN  = 0;
constexpr size_t OFF_QKV = OFF_XN + SZ_XN;
constexpr size_t OFF_Q16 = OFF_QKV + SZ_QKV;
constexpr size_t OFF_K16 = OFF_Q16 + SZ_Q16;
constexpr size_t OFF_V16 = OFF_K16 + SZ_Q16;
constexpr size_t OFF_BIAS= OFF_V16 + SZ_Q16;                       // f32 [12,128,128]
constexpr size_t OFF_SC1 = OFF_BIAS + (size_t)kHE * kH * kW * 4;
constexpr size_t OFF_SC2 = OFF_SC1 + (size_t)kB * kC * 4;
constexpr size_t OFF_SC3 = OFF_SC2 + (size_t)kB * kC * 4;
constexpr size_t OFF_INW = (OFF_SC3 + (size_t)kB * kC * 4 + 255) & ~(size_t)255;
constexpr size_t OFF_OUTW= OFF_INW + (size_t)kC3 * kC * 2;
constexpr size_t OFF_W1T = OFF_OUTW + (size_t)kC * kC * 2;
constexpr size_t OFF_W2T = OFF_W1T + (size_t)kC4 * kC * 2;
// aliases of dead regions
constexpr size_t OFF_ATTN= OFF_QKV;                    // f32 [M,C]   (qkv dead)
constexpr size_t OFF_RES2= OFF_QKV;                    // f32 [M,C]   (attn dead)
constexpr size_t OFF_T   = OFF_QKV + (size_t)kM*kC*4;  // f16 [M,4C]  (rest of qkv)
constexpr size_t OFF_X216= OFF_Q16;                    // f16 [M,C]   (q16 dead)
constexpr size_t OFF_H   = OFF_Q16 + SZ_Q16;           // f32 [M,C]   (k16/v16 dead)
} // namespace

__device__ __forceinline__ v8f wmma16(v16h a, v16h b, v8f c) {
  return __builtin_amdgcn_wmma_f32_16x16x32_f16(false, a, false, b, (short)0, c,
                                                false, false);
}

// ---- Tensor Data Mover: 2D tile load (rows x cols f16, row stride in elems)
// D# per CDNA5 ISA: group0 = {count/flags, lds_addr,
// global_addr lo, global_addr hi | type=2}; group1 packs data_size=2B,
// tensor_dim0/1, tile_dim0/1, tensor_dim0_stride. Tracked by TENSORcnt.
__device__ __forceinline__ void tdm_load_2d_f16(unsigned lds_addr,
                                                const _Float16* gptr,
                                                int rows, int cols,
                                                int row_stride) {
  unsigned long long ga = (unsigned long long)(size_t)gptr;
  u32x4 g0;
  g0[0] = 1u;                                          // count=1, no gather
  g0[1] = lds_addr;                                    // LDS byte address
  g0[2] = (unsigned)(ga & 0xffffffffu);                // global_addr[31:0]
  g0[3] = (unsigned)((ga >> 32) & 0x01ffffffu) | (2u << 30);  // addr hi, type=2
  i32x8 g1;
  g1[0] = 1 << 16;                                     // data_size=1 (2 bytes)
  g1[1] = (cols & 0xffff) << 16;                       // tensor_dim0[15:0]
  g1[2] = ((cols >> 16) & 0xffff) | ((rows & 0xffff) << 16);  // dim0 hi | dim1 lo
  g1[3] = ((rows >> 16) & 0xffff) | ((cols & 0xffff) << 16);  // dim1 hi | tile_dim0
  g1[4] = rows & 0xffff;                               // tile_dim1 (tile_dim2=0)
  g1[5] = row_stride;                                  // tensor_dim0_stride[31:0]
  g1[6] = 0;                                           // stride hi | dim1_stride lo
  g1[7] = 0;
  i32x4 z4 = {0, 0, 0, 0};
#if defined(__clang_major__) && (__clang_major__ >= 23)
  i32x8 z8 = {0, 0, 0, 0, 0, 0, 0, 0};
  __builtin_amdgcn_tensor_load_to_lds(g0, g1, z4, z4, z8, 0);
#else
  __builtin_amdgcn_tensor_load_to_lds(g0, g1, z4, z4, 0);
#endif
}

// ---------- per-(b,c) unbiased std over HW -> scale = w[c]/(std+eps) ----------
__global__ __launch_bounds__(256) void k_std_scales(
    const float* __restrict__ x, float* __restrict__ scales,
    const float* __restrict__ w, int chw_layout) {
  int bc = blockIdx.x;
  int b = bc / kC, c = bc % kC;
  const float* p; size_t stride;
  if (chw_layout) { p = x + (size_t)bc * kHW; stride = 1; }
  else            { p = x + (size_t)b * kHW * kC + c; stride = (size_t)kC; }
  float s = 0.f, ss = 0.f;
  for (int i = threadIdx.x; i < kHW; i += 256) {
    float v = p[(size_t)i * stride];
    s += v; ss += v * v;
  }
  __shared__ float sb[256], qb[256];
  sb[threadIdx.x] = s; qb[threadIdx.x] = ss;
  __syncthreads();
  for (int o = 128; o > 0; o >>= 1) {
    if ((int)threadIdx.x < o) {
      sb[threadIdx.x] += sb[threadIdx.x + o];
      qb[threadIdx.x] += qb[threadIdx.x + o];
    }
    __syncthreads();
  }
  if (threadIdx.x == 0) {
    float n = (float)kHW;
    float var = (qb[0] - sb[0] * sb[0] / n) / (n - 1.f);
    float sd = sqrtf(fmaxf(var, 0.f));
    scales[bc] = w[c] / (sd + kEpsIn);
  }
}

// ---------- normalize x [B,C,HW] -> xn f16 [M=BHW, C] (transpose) ----------
__global__ void k_norm_x_to_f16(const float* __restrict__ x,
                                const float* __restrict__ scales,
                                _Float16* __restrict__ xn) {
  size_t idx = (size_t)blockIdx.x * 256 + threadIdx.x;   // over M*C
  int c = (int)(idx % kC);
  size_t m = idx / kC;
  int b = (int)(m / kHW);
  int pos = (int)(m % kHW);
  float v = x[((size_t)b * kC + c) * kHW + pos] * scales[b * kC + c];
  xn[idx] = (_Float16)v;
}

// ---------- normalize [M,C] fp32 -> f16 ----------
__global__ void k_norm_pc_to_f16(const float* __restrict__ src,
                                 const float* __restrict__ scales,
                                 _Float16* __restrict__ dst) {
  size_t idx = (size_t)blockIdx.x * 256 + threadIdx.x;
  int c = (int)(idx % kC);
  size_t m = idx / kC;
  int b = (int)(m / kHW);
  dst[idx] = (_Float16)(src[idx] * scales[b * kC + c]);
}

__global__ void k_f32_to_f16(const float* __restrict__ in,
                             _Float16* __restrict__ out, int n) {
  int i = blockIdx.x * 256 + threadIdx.x;
  if (i < n) out[i] = (_Float16)in[i];
}

// in [K,N] f32 -> out [N,K] f16
__global__ void k_transpose_to_f16(const float* __restrict__ in,
                                   _Float16* __restrict__ out, int K, int N) {
  int i = blockIdx.x * 256 + threadIdx.x;
  if (i < K * N) {
    int n = i / K, k = i % K;
    out[i] = (_Float16)in[(size_t)k * N + n];
  }
}

// ---------- T5 relative bias table bias[head][i][j] ----------
__global__ void k_bias_table(const float* __restrict__ emb,
                             float* __restrict__ bias) {
  int idx = blockIdx.x * 256 + threadIdx.x;       // 12*128*128
  int h = idx / (kH * kW);
  int rem = idx % (kH * kW);
  int i = rem / kW, j = rem % kW;
  int rp = j - i;
  int ret = rp > 0 ? 16 : 0;
  int arp = rp < 0 ? -rp : rp;
  int bucket;
  if (arp < 8) {
    bucket = ret + arp;
  } else {
    float rpf = (float)arp;
    int vl = 8 + (int)(logf(rpf * 0.125f) * (1.0f / logf(16.0f)) * 8.0f);
    if (vl > 15) vl = 15;
    bucket = ret + vl;
  }
  bias[idx] = emb[bucket * kHE + h];
}

// ---------- qkv f32 [M,3C] -> LN(q),LN(k),v f16 in [B,he,H,W,hd] ----------
__global__ __launch_bounds__(256) void k_qkv_prep(
    const float* __restrict__ qkv,
    const float* __restrict__ qn_w, const float* __restrict__ qn_b,
    const float* __restrict__ kn_w, const float* __restrict__ kn_b,
    _Float16* __restrict__ q16, _Float16* __restrict__ k16,
    _Float16* __restrict__ v16) {
  int gid = blockIdx.x * 8 + (threadIdx.x >> 5);
  int lane = threadIdx.x & 31;
  int head = gid % kHE;
  int bp = gid / kHE;
  int pos = bp % kHW;
  int b = bp / kHW;
  const float* src = qkv + (size_t)(b * kHW + pos) * kC3 + head * (3 * kHD);
  size_t dst = ((size_t)(b * kHE + head) * kHW + pos) * kHD;
  int d0 = lane, d1 = lane + 32;
  {
    float a = src[d0], bb = src[d1];
    float s = a + bb;
    for (int m = 16; m >= 1; m >>= 1) s += __shfl_xor(s, m, 32);
    float mean = s * (1.0f / 64.0f);
    float da = a - mean, db = bb - mean;
    float vv = da * da + db * db;
    for (int m = 16; m >= 1; m >>= 1) vv += __shfl_xor(vv, m, 32);
    float inv = rsqrtf(vv * (1.0f / 64.0f) + kEpsLn);
    q16[dst + d0] = (_Float16)(da * inv * qn_w[d0] + qn_b[d0]);
    q16[dst + d1] = (_Float16)(db * inv * qn_w[d1] + qn_b[d1]);
  }
  {
    const float* sk = src + kHD;
    float a = sk[d0], bb = sk[d1];
    float s = a + bb;
    for (int m = 16; m >= 1; m >>= 1) s += __shfl_xor(s, m, 32);
    float mean = s * (1.0f / 64.0f);
    float da = a - mean, db = bb - mean;
    float vv = da * da + db * db;
    for (int m = 16; m >= 1; m >>= 1) vv += __shfl_xor(vv, m, 32);
    float inv = rsqrtf(vv * (1.0f / 64.0f) + kEpsLn);
    k16[dst + d0] = (_Float16)(da * inv * kn_w[d0] + kn_b[d0]);
    k16[dst + d1] = (_Float16)(db * inv * kn_w[d1] + kn_b[d1]);
  }
  v16[dst + d0] = (_Float16)src[2 * kHD + d0];
  v16[dst + d1] = (_Float16)src[2 * kHD + d1];
}

// ---------- WMMA GEMM: C[m,n] = sum_k A[m,k]*Bm[n,k] + bias[n]  ----------
// 128x128 block tile, BK=32, 8 waves -> each wave 32x64 (2x4 wmma tiles).
// A/B tiles staged into double-buffered LDS by the Tensor Data Mover (wave 0
// issues; TENSORcnt-synchronized, 2-tile pipeline depth).
// EPI: 0=store f32 (qkv)  1=out-proj(gamma,residual; f32+f16 out)
//      2=GELU -> f16      3=store f32 (mlp2)
template <int EPI>
__global__ __launch_bounds__(256) void k_gemm(
    const _Float16* __restrict__ A, const _Float16* __restrict__ Bm,
    int M, int N, int K,
    const float* __restrict__ bias,
    float* __restrict__ outF, _Float16* __restrict__ outH,
    const float* __restrict__ gamma, const float* __restrict__ resid) {
  __shared__ _Float16 As[2][128][32];   // 16 KB
  __shared__ _Float16 Bs[2][128][32];   // 16 KB
  const int t = threadIdx.x;
  const int m0 = blockIdx.y * 128;
  const int n0 = blockIdx.x * 128;
  const int lane = t & 31;
  const int w = t >> 5;
  const int wm = w & 3;        // 32-row group
  const int wn = w >> 2;       // 64-col group
  const int fr = lane & 15;
  const int kh = (lane >> 4) << 4;
  const int nk = K / 32;
  const _Float16* Abase = A + (size_t)m0 * K;
  const _Float16* Bbase = Bm + (size_t)n0 * K;
  const unsigned ldsA[2] = {(unsigned)(size_t)&As[0][0][0],
                            (unsigned)(size_t)&As[1][0][0]};
  const unsigned ldsB[2] = {(unsigned)(size_t)&Bs[0][0][0],
                            (unsigned)(size_t)&Bs[1][0][0]};
  if (w == 0) {                         // prime 2-deep TDM pipeline
    tdm_load_2d_f16(ldsA[0], Abase, 128, 32, K);
    tdm_load_2d_f16(ldsB[0], Bbase, 128, 32, K);
    if (nk > 1) {
      tdm_load_2d_f16(ldsA[1], Abase + 32, 128, 32, K);
      tdm_load_2d_f16(ldsB[1], Bbase + 32, 128, 32, K);
    }
  }
  v8f acc[2][4] = {};
  for (int kt = 0; kt < nk; ++kt) {
    const int cur = kt & 1;
    if (w == 0) {                       // wait for tile kt (next stays in flight)
      if (kt + 1 < nk) __builtin_amdgcn_s_wait_tensorcnt(2);
      else             __builtin_amdgcn_s_wait_tensorcnt(0);
    }
    __syncthreads();                    // tile kt visible to all waves
    v16h af[2], bf[4];
#pragma unroll
    for (int i = 0; i < 2; ++i)
      af[i] = *(const v16h*)&As[cur][wm * 32 + i * 16 + fr][kh];
#pragma unroll
    for (int j = 0; j < 4; ++j)
      bf[j] = *(const v16h*)&Bs[cur][wn * 64 + j * 16 + fr][kh];
#pragma unroll
    for (int i = 0; i < 2; ++i)
#pragma unroll
      for (int j = 0; j < 4; ++j)
        acc[i][j] = wmma16(af[i], bf[j], acc[i][j]);
    __syncthreads();                    // all waves done reading buf `cur`
    if (w == 0 && kt + 2 < nk) {        // refill freed buffer
      tdm_load_2d_f16(ldsA[cur], Abase + (kt + 2) * 32, 128, 32, K);
      tdm_load_2d_f16(ldsB[cur], Bbase + (kt + 2) * 32, 128, 32, K);
    }
  }
#pragma unroll
  for (int i = 0; i < 2; ++i) {
#pragma unroll
    for (int j = 0; j < 4; ++j) {
      const int mg = m0 + wm * 32 + i * 16 + (lane >> 4) * 8;
      const int ng = n0 + wn * 64 + j * 16 + (lane & 15);
      const float bv = bias[ng];
#pragma unroll
      for (int r = 0; r < 8; ++r) {
        const int m = mg + r;
        float v = acc[i][j][r] + bv;
        size_t oi = (size_t)m * N + ng;
        if (EPI == 0) {
          outF[oi] = v;
        } else if (EPI == 1) {
          int b = m / kHW, pos = m % kHW;
          float r0 = resid[((size_t)b * N + ng) * kHW + pos];
          float x2 = v * gamma[ng] + r0;
          outF[oi] = x2;
          outH[oi] = (_Float16)x2;
        } else if (EPI == 2) {
          float g = 0.5f * v * (1.0f + erff(v * 0.70710678118f));
          outH[oi] = (_Float16)g;
        } else {
          outF[oi] = v;
        }
      }
    }
  }
}

// ---------- axial attention: one block per (b, head, line) ----------
// S = (Q K^T)/8 + bias ; P = softmax(S) ; O = P V ; out += / = 0.5*O
__global__ __launch_bounds__(256) void k_attn(
    const _Float16* __restrict__ q16, const _Float16* __restrict__ k16,
    const _Float16* __restrict__ v16, const float* __restrict__ bias,
    float* __restrict__ attn_out, int is_col) {
  __shared__ _Float16 Ks[128][64];       // 16 KB
  __shared__ _Float16 Vt[64][128];       // 16 KB (V transposed: [d][pos])
  __shared__ _Float16 Ps[8][16][128];    // 32 KB (per-wave P)
  const int t = threadIdx.x;
  const int bidx = blockIdx.x;
  const int line = bidx & 127;
  const int bh = bidx >> 7;
  const int head = bh % kHE;
  const int b = bh / kHE;
  size_t base, sstr;
  if (is_col) {
    base = ((size_t)(b * kHE + head) * kHW + line) * kHD;
    sstr = (size_t)kW * kHD;
  } else {
    base = ((size_t)((b * kHE + head) * kH + line) * kW) * kHD;
    sstr = (size_t)kHD;
  }
  { // stage K rows and V^T
    int row = t >> 1;
    int c0 = (t & 1) * 32;
    const _Float16* kp = k16 + base + (size_t)row * sstr + c0;
    *(v16h*)&Ks[row][c0]      = *(const v16h*)(kp);
    *(v16h*)&Ks[row][c0 + 16] = *(const v16h*)(kp + 16);
    const _Float16* vp = v16 + base + (size_t)row * sstr + c0;
#pragma unroll
    for (int i = 0; i < 32; ++i) Vt[c0 + i][row] = vp[i];
  }
  __syncthreads();
  const int w = t >> 5;
  const int lane = t & 31;
  const int fr = lane & 15;
  const int kh = (lane >> 4) << 4;
  const int qr0 = w * 16;                  // this wave's 16 query rows
  const _Float16* qrow = q16 + base + (size_t)(qr0 + fr) * sstr;
  v16h aq0 = *(const v16h*)(qrow + kh);
  v16h aq1 = *(const v16h*)(qrow + 32 + kh);
  v8f s[8];
#pragma unroll
  for (int j = 0; j < 8; ++j) {
    v8f z = {};
    v16h bk0 = *(const v16h*)&Ks[j * 16 + fr][kh];
    v16h bk1 = *(const v16h*)&Ks[j * 16 + fr][32 + kh];
    z = wmma16(aq0, bk0, z);
    z = wmma16(aq1, bk1, z);
    s[j] = z;
  }
  const int half = lane >> 4;
  const int nl = lane & 15;
#pragma unroll
  for (int r = 0; r < 8; ++r) {
    const int ml = r + half * 8;
    const int mq = qr0 + ml;
    const float* bp = bias + ((size_t)head * kH + mq) * kW;
    float rmax = -3.4e38f;
#pragma unroll
    for (int j = 0; j < 8; ++j) {
      float v = s[j][r] * 0.125f + bp[j * 16 + nl];
      s[j][r] = v;
      rmax = fmaxf(rmax, v);
    }
    rmax = fmaxf(rmax, __shfl_xor(rmax, 1, 32));
    rmax = fmaxf(rmax, __shfl_xor(rmax, 2, 32));
    rmax = fmaxf(rmax, __shfl_xor(rmax, 4, 32));
    rmax = fmaxf(rmax, __shfl_xor(rmax, 8, 32));
    float rsum = 0.f;
#pragma unroll
    for (int j = 0; j < 8; ++j) {
      float e = __expf(s[j][r] - rmax);
      s[j][r] = e;
      rsum += e;
    }
    rsum += __shfl_xor(rsum, 1, 32);
    rsum += __shfl_xor(rsum, 2, 32);
    rsum += __shfl_xor(rsum, 4, 32);
    rsum += __shfl_xor(rsum, 8, 32);
    float rinv = 1.0f / rsum;
#pragma unroll
    for (int j = 0; j < 8; ++j)
      Ps[w][ml][j * 16 + nl] = (_Float16)(s[j][r] * rinv);
  }
  __syncthreads();
  v8f o[4] = {};
#pragma unroll
  for (int ks = 0; ks < 4; ++ks) {
    v16h ap = *(const v16h*)&Ps[w][fr][ks * 32 + kh];
#pragma unroll
    for (int jd = 0; jd < 4; ++jd) {
      v16h bv = *(const v16h*)&Vt[jd * 16 + fr][ks * 32 + kh];
      o[jd] = wmma16(ap, bv, o[jd]);
    }
  }
#pragma unroll
  for (int jd = 0; jd < 4; ++jd) {
    const int d = jd * 16 + nl;
    const int c = head * kHD + d;
#pragma unroll
    for (int r = 0; r < 8; ++r) {
      const int mq = qr0 + r + half * 8;
      const int pos = is_col ? (mq * kW + line) : (line * kW + mq);
      size_t oi = ((size_t)b * kHW + pos) * kC + c;
      float v = o[jd][r] * 0.5f;
      if (is_col) attn_out[oi] += v;
      else attn_out[oi] = v;
    }
  }
}

// ---------- final: out[b,c,h,w] = res2 + gamma_mlp[c]*h*scale3 ----------
__global__ void k_final(const float* __restrict__ res2,
                        const float* __restrict__ h,
                        const float* __restrict__ sc3,
                        const float* __restrict__ gamma_mlp,
                        float* __restrict__ out) {
  size_t idx = (size_t)blockIdx.x * 256 + threadIdx.x;     // B*C*HW
  int b = (int)(idx / ((size_t)kC * kHW));
  int rem = (int)(idx % ((size_t)kC * kHW));
  int c = rem / kHW;
  int pos = rem % kHW;
  size_t pi = ((size_t)b * kHW + pos) * kC + c;
  out[idx] = res2[pi] + gamma_mlp[c] * h[pi] * sc3[b * kC + c];
}

extern "C" void kernel_launch(void* const* d_in, const int* in_sizes, int n_in,
                              void* d_out, int out_size, void* d_ws, size_t ws_size,
                              hipStream_t stream) {
  const float* x        = (const float*)d_in[0];
  const float* norm1_w  = (const float*)d_in[2];
  const float* norm2_w  = (const float*)d_in[3];
  const float* mlpnorm_w= (const float*)d_in[4];
  const float* gamma_att= (const float*)d_in[5];
  const float* gamma_mlp= (const float*)d_in[6];
  const float* in_w     = (const float*)d_in[7];
  const float* in_b     = (const float*)d_in[8];
  const float* out_w    = (const float*)d_in[9];
  const float* out_b    = (const float*)d_in[10];
  const float* qn_w     = (const float*)d_in[11];
  const float* qn_b     = (const float*)d_in[12];
  const float* kn_w     = (const float*)d_in[13];
  const float* kn_b     = (const float*)d_in[14];
  const float* emb      = (const float*)d_in[15];
  const float* mlp_w1   = (const float*)d_in[16];
  const float* mlp_b1   = (const float*)d_in[17];
  const float* mlp_w2   = (const float*)d_in[18];
  const float* mlp_b2   = (const float*)d_in[19];

  char* ws = (char*)d_ws;
  _Float16* xn    = (_Float16*)(ws + OFF_XN);
  _Float16* xn2   = (_Float16*)(ws + OFF_XN);     // alias (xn dead)
  float*    qkv   = (float*)(ws + OFF_QKV);
  float*    attnO = (float*)(ws + OFF_ATTN);      // alias (qkv dead)
  float*    res2  = (float*)(ws + OFF_RES2);      // alias (attnO dead)
  _Float16* tbuf  = (_Float16*)(ws + OFF_T);
  _Float16* q16b  = (_Float16*)(ws + OFF_Q16);
  _Float16* k16b  = (_Float16*)(ws + OFF_K16);
  _Float16* v16b  = (_Float16*)(ws + OFF_V16);
  _Float16* x216  = (_Float16*)(ws + OFF_X216);   // alias (q16 dead)
  float*    hbuf  = (float*)(ws + OFF_H);         // alias (k16/v16 dead)
  float*    biasT = (float*)(ws + OFF_BIAS);
  float*    sc1   = (float*)(ws + OFF_SC1);
  float*    sc2   = (float*)(ws + OFF_SC2);
  float*    sc3   = (float*)(ws + OFF_SC3);
  _Float16* inw16 = (_Float16*)(ws + OFF_INW);
  _Float16* outw16= (_Float16*)(ws + OFF_OUTW);
  _Float16* w1T   = (_Float16*)(ws + OFF_W1T);
  _Float16* w2T   = (_Float16*)(ws + OFF_W2T);

  // weight prep (f16, B operands stored [N,K])
  k_f32_to_f16<<<(kC3 * kC + 255) / 256, 256, 0, stream>>>(in_w, inw16, kC3 * kC);
  k_f32_to_f16<<<(kC * kC + 255) / 256, 256, 0, stream>>>(out_w, outw16, kC * kC);
  k_transpose_to_f16<<<(kC * kC4 + 255) / 256, 256, 0, stream>>>(mlp_w1, w1T, kC, kC4);
  k_transpose_to_f16<<<(kC4 * kC + 255) / 256, 256, 0, stream>>>(mlp_w2, w2T, kC4, kC);
  k_bias_table<<<kHE * kH * kW / 256, 256, 0, stream>>>(emb, biasT);

  // rms_inorm(x) -> xn f16 [M,C]
  k_std_scales<<<kB * kC, 256, 0, stream>>>(x, sc1, norm1_w, 1);
  k_norm_x_to_f16<<<kM * (kC / 256), 256, 0, stream>>>(x, sc1, xn);

  // qkv projection
  k_gemm<0><<<dim3(kC3 / 128, kM / 128), 256, 0, stream>>>(
      xn, inw16, kM, kC3, kC, in_b, qkv, nullptr, nullptr, nullptr);

  // layernorm q/k + f16 split
  k_qkv_prep<<<kB * kHW * kHE / 8, 256, 0, stream>>>(
      qkv, qn_w, qn_b, kn_w, kn_b, q16b, k16b, v16b);

  // row then column attention (col accumulates)
  k_attn<<<kB * kHE * kH, 256, 0, stream>>>(q16b, k16b, v16b, biasT, attnO, 0);
  k_attn<<<kB * kHE * kW, 256, 0, stream>>>(q16b, k16b, v16b, biasT, attnO, 1);

  // rms_inorm(attn) -> xn2 f16
  k_std_scales<<<kB * kC, 256, 0, stream>>>(attnO, sc2, norm2_w, 0);
  k_norm_pc_to_f16<<<kM * (kC / 256), 256, 0, stream>>>(attnO, sc2, xn2);

  // out projection + gamma + residual -> res2 f32, x2 f16
  k_gemm<1><<<dim3(kC / 128, kM / 128), 256, 0, stream>>>(
      xn2, outw16, kM, kC, kC, out_b, res2, x216, gamma_att, x);

  // MLP
  k_gemm<2><<<dim3(kC4 / 128, kM / 128), 256, 0, stream>>>(
      x216, w1T, kM, kC4, kC, mlp_b1, nullptr, tbuf, nullptr, nullptr);
  k_gemm<3><<<dim3(kC / 128, kM / 128), 256, 0, stream>>>(
      tbuf, w2T, kM, kC, kC4, mlp_b2, hbuf, nullptr, nullptr, nullptr);

  // rms_inorm(mlp) + residual -> out [B,C,H,W]
  k_std_scales<<<kB * kC, 256, 0, stream>>>(hbuf, sc3, mlpnorm_w, 0);
  k_final<<<kM * (kC / 256), 256, 0, stream>>>(res2, hbuf, sc3, gamma_mlp,
                                               (float*)d_out);
}
